// CircA_1271310320121
// MI455X (gfx1250) — compile-verified, matrix operand order checked by reading, not run on hardware
//
#include <hip/hip_runtime.h>
#include <cstdint>
#include <cstddef>

// ---------------------------------------------------------------------------
// CircA: y = (X * psi) circ-conv g[r], weighted by omega[r], summed over r.
// Collapses to Y[M,N] = Xh[M,K] * Wn[N,K]^T  with
//   Wn[n][k] = psi[k] * sum_r omega[r,n] * g[r, (n-k) mod 4096]
// M = 8192, N = K = 4096.  f16 inputs, f32 accumulate via v_wmma_f32_16x16x32_f16.
// Staging: Tensor Data Mover (tensor_load_to_lds, 6-arg clang-23 form) if
// available, else global_load_async_to_lds_b128.
// ---------------------------------------------------------------------------

#if defined(__has_builtin)
#if __has_builtin(__builtin_amdgcn_tensor_load_to_lds)
#define CIRCA_TDM 1
#endif
#endif

typedef __attribute__((ext_vector_type(16))) _Float16 v16h;
typedef __attribute__((ext_vector_type(8)))  float    v8f;
typedef __attribute__((ext_vector_type(4)))  _Float16 v4h;
typedef unsigned int u32x4 __attribute__((ext_vector_type(4)));
typedef int          i32x8 __attribute__((ext_vector_type(8)));
typedef int          i32x4 __attribute__((ext_vector_type(4)));

union Frag { v16h h; uint32_t u[8]; };

#define KDIM 4096
#define NDIM 4096
#define BK   32
#define BM   128
#define BN   256
#define ROW_BYTES 80            // 32 halfs (64B) padded to 80B -> 20 banks, conflict-free
#define ROW_U32   20
#define A_TILE_BYTES (BM * ROW_BYTES)            // 10240
#define B_TILE_BYTES (BN * ROW_BYTES)            // 20480
#define BUF_BYTES    (A_TILE_BYTES + B_TILE_BYTES) // 30720

// ------------------------- TDM descriptor issue -----------------------------
#if CIRCA_TDM
// 2-D tile load: rows x 32 f16 elements, row stride 4096 elements in memory,
// LDS rows padded 64B -> 80B via pad_interval(16 dwords)/pad_amount(4 dwords).
__device__ __forceinline__ void tdm_load_tile(uint32_t lds_addr, uint64_t gaddr,
                                              int rows) {
  u32x4 g0;
  g0[0] = 1u;                                            // count=1, user mode
  g0[1] = lds_addr;                                      // LDS byte address
  g0[2] = (uint32_t)gaddr;                               // global addr [31:0]
  g0[3] = (uint32_t)((gaddr >> 32) & 0x1FFFFFFu)         // global addr [56:32]
          | (2u << 30);                                  // type = 2 (image)
  i32x8 g1;
  g1[0] = (int)((1u << 16)        // data_size = 1 -> 2 bytes
              | (1u << 20)        // pad_enable
              | (3u << 22)        // pad_interval: 16 dwords (64B)
              | (3u << 25));      // pad_amount:   4 dwords (16B)
  g1[1] = (int)(4096u << 16);     // tensor_dim0 = 4096 (bits 79:48, low16 here)
  g1[2] = (int)((uint32_t)rows << 16); // tensor_dim0 hi=0 | tensor_dim1 low16
  g1[3] = (int)(32u << 16);       // tensor_dim1 hi=0 | tile_dim0 = 32 elems
  g1[4] = rows;                   // tile_dim1 = rows, tile_dim2 = 0
  g1[5] = 4096;                   // tensor_dim0_stride (elements), low 32
  g1[6] = 0;
  g1[7] = 0;
  i32x4 zz4 = {0, 0, 0, 0};
  i32x8 zz8 = {0, 0, 0, 0, 0, 0, 0, 0};
  // clang-23 / therock form: (g0, g1, g2, g3, extra, cpol)
  __builtin_amdgcn_tensor_load_to_lds(g0, g1, zz4, zz4, zz8, 0);
}
__device__ __forceinline__ void wait_tdm() {
  __builtin_amdgcn_s_wait_tensorcnt((short)0);
}
#else
// ----------------- fallback: per-lane async global -> LDS -------------------
__device__ __forceinline__ void stage16(void* lptr, const void* gptr) {
  asm volatile("global_load_async_to_lds_b128 %0, %1, off"
               :: "v"((uint32_t)(uintptr_t)lptr),
                  "v"((unsigned long long)(uintptr_t)gptr)
               : "memory");
}
__device__ __forceinline__ void wait_stage() {
#if __has_builtin(__builtin_amdgcn_s_wait_asynccnt)
  __builtin_amdgcn_s_wait_asynccnt(0);
#else
  asm volatile("s_wait_asynccnt 0" ::: "memory");
#endif
}
#endif

// --------------------------- kernel 1: X fp32 -> f16 ------------------------
__global__ void circa_convert_x(const float* __restrict__ x,
                                _Float16* __restrict__ xh, int total4) {
  int i = blockIdx.x * blockDim.x + threadIdx.x;
  if (i < total4) {
    float4 v = ((const float4*)x)[i];
    v4h h = { (_Float16)v.x, (_Float16)v.y, (_Float16)v.z, (_Float16)v.w };
    ((v4h*)xh)[i] = h;
  }
}

// --------------------------- kernel 2: build Wn -----------------------------
__global__ void circa_build_w(const float* __restrict__ psi,
                              const float* __restrict__ omega,
                              const float* __restrict__ g,
                              _Float16* __restrict__ Wn) {
  int idx = blockIdx.x * blockDim.x + threadIdx.x;   // idx = n*4096 + k
  int n = idx >> 12;
  int k = idx & 4095;
  float acc = 0.f;
#pragma unroll
  for (int r = 0; r < 3; ++r)
    acc += omega[r * 4096 + n] * g[r * 4096 + ((n - k) & 4095)];
  Wn[idx] = (_Float16)(psi[k] * acc);
}

// --------------------------- kernel 3: WMMA GEMM ----------------------------
// Block = 256 threads = 8 waves; tile 128(M) x 256(N); waves 2(M) x 4(N),
// each wave owns 64x64 = 4x4 subtiles of 16x16 -> 16 WMMA per K-step.
__global__ __launch_bounds__(256)
void circa_gemm(const _Float16* __restrict__ Xh,
                const _Float16* __restrict__ Wn,
                const float* __restrict__ bias,
                float* __restrict__ Y) {
  __shared__ __align__(16) unsigned char smem[2 * BUF_BYTES];

  const int tid   = threadIdx.x;
  const int lane  = tid & 31;
  const int wave  = tid >> 5;
  const int waveM = wave >> 2;        // 0..1 -> 64 rows each
  const int waveN = wave & 3;         // 0..3 -> 64 cols each
  const int row16 = lane & 15;
  const int hi    = (lane >> 4) & 1;
  const int m0    = blockIdx.y * BM;
  const int n0    = blockIdx.x * BN;

  v8f acc[4][4] = {};

#if CIRCA_TDM
  // Waves 0 and 1 drive the Tensor Data Mover (EXEC-independent, wave-level).
  auto issue = [&](int buf, int kt) {
    const int kb = kt * BK;
    const uint32_t base = (uint32_t)(uintptr_t)(&smem[0]) + buf * BUF_BYTES;
    if (wave == 0)
      tdm_load_tile(base,
                    (uint64_t)(uintptr_t)(Xh + (size_t)m0 * KDIM + kb), BM);
    else if (wave == 1)
      tdm_load_tile(base + A_TILE_BYTES,
                    (uint64_t)(uintptr_t)(Wn + (size_t)n0 * KDIM + kb), BN);
  };
#else
  auto issue = [&](int buf, int kt) {
    const int kb = kt * BK;
    unsigned char* base = smem + buf * BUF_BYTES;
#pragma unroll
    for (int i = 0; i < 2; ++i) {     // A: 128 rows x 4 chunks = 512
      int id = tid + i * 256, row = id >> 2, c = id & 3;
      stage16(base + row * ROW_BYTES + c * 16,
              Xh + (size_t)(m0 + row) * KDIM + kb + c * 8);
    }
#pragma unroll
    for (int i = 0; i < 4; ++i) {     // B: 256 rows x 4 chunks = 1024
      int id = tid + i * 256, row = id >> 2, c = id & 3;
      stage16(base + A_TILE_BYTES + row * ROW_BYTES + c * 16,
              Wn + (size_t)(n0 + row) * KDIM + kb + c * 8);
    }
  };
#endif

  issue(0, 0);

  const int NT = KDIM / BK;           // 128
  for (int kt = 0; kt < NT; ++kt) {
    const int cur = kt & 1;
#if CIRCA_TDM
    if (wave < 2) wait_tdm();
#else
    wait_stage();
#endif
    __syncthreads();
    if (kt + 1 < NT) issue(cur ^ 1, kt + 1);

    const uint32_t* As = (const uint32_t*)(smem + cur * BUF_BYTES);
    const uint32_t* Bs = (const uint32_t*)(smem + cur * BUF_BYTES + A_TILE_BYTES);

    // Fragment layout (16-bit WMMA, §7.12.2): lane = hi*16 + row16,
    // VGPR v holds K pair k0 = (v>=4?16:0) + hi*8 + 2*(v&3).
    Frag b[4];
#pragma unroll
    for (int ni = 0; ni < 4; ++ni) {
      const int col = waveN * 64 + ni * 16 + row16;
#pragma unroll
      for (int v = 0; v < 8; ++v) {
        const int u = (v & 3) + hi * 4 + ((v >> 2) << 3);
        b[ni].u[v] = Bs[col * ROW_U32 + u];
      }
    }
#pragma unroll
    for (int mi = 0; mi < 4; ++mi) {
      Frag a;
      const int arow = waveM * 64 + mi * 16 + row16;
#pragma unroll
      for (int v = 0; v < 8; ++v) {
        const int u = (v & 3) + hi * 4 + ((v >> 2) << 3);
        a.u[v] = As[arow * ROW_U32 + u];
      }
#pragma unroll
      for (int ni = 0; ni < 4; ++ni)
        acc[mi][ni] = __builtin_amdgcn_wmma_f32_16x16x32_f16(
            false, a.h, false, b[ni].h, (short)0, acc[mi][ni], false, false);
    }
    __syncthreads();
  }

  // Epilogue: C layout — VGPR v, lanes0-15: M=v, lanes16-31: M=8+v; N=row16.
#pragma unroll
  for (int ni = 0; ni < 4; ++ni) {
    const int n  = n0 + waveN * 64 + ni * 16 + row16;
    const float bv = bias[n];
#pragma unroll
    for (int mi = 0; mi < 4; ++mi) {
      const int mbase = m0 + waveM * 64 + mi * 16 + hi * 8;
#pragma unroll
      for (int v = 0; v < 8; ++v)
        Y[(size_t)(mbase + v) * NDIM + n] = acc[mi][ni][v] + bv;
    }
  }
}

// ---------------------------------------------------------------------------
extern "C" void kernel_launch(void* const* d_in, const int* in_sizes, int n_in,
                              void* d_out, int out_size, void* d_ws, size_t ws_size,
                              hipStream_t stream) {
  const float* x     = (const float*)d_in[0];   // (4,2048,4096) -> M x K
  const float* psi   = (const float*)d_in[1];   // (1,4096)
  const float* omega = (const float*)d_in[2];   // (3,4096)
  const float* g     = (const float*)d_in[3];   // (3,4096)
  const float* bias  = (const float*)d_in[4];   // (4096,)
  float* Y = (float*)d_out;

  const int M = in_sizes[0] / KDIM;             // 8192

  _Float16* Xh = (_Float16*)d_ws;                                 // M*K f16
  _Float16* Wn = (_Float16*)((char*)d_ws + (size_t)M * KDIM * 2); // N*K f16

  // 1) X -> f16
  {
    int total4 = (M * KDIM) / 4;
    circa_convert_x<<<(total4 + 255) / 256, 256, 0, stream>>>(x, Xh, total4);
  }
  // 2) Build structured weight matrix Wn[n][k]
  {
    int total = NDIM * KDIM;
    circa_build_w<<<total / 256, 256, 0, stream>>>(psi, omega, g, Wn);
  }
  // 3) GEMM  Y[M,N] = Xh[M,K] * Wn[N,K]^T + bias
  {
    dim3 grid(NDIM / BN, M / BM);
    circa_gemm<<<grid, 256, 0, stream>>>(Xh, Wn, bias, Y);
  }
}
